// SimpleLESS4FDModel_12824772345882
// MI455X (gfx1250) — compile-verified, matrix operand order checked by reading, not run on hardware
//
#include <hip/hip_runtime.h>

typedef __attribute__((ext_vector_type(16))) _Float16 v16h;
typedef __attribute__((ext_vector_type(8)))  float    v8f;

#define NNODE 4096
#define CDIM  256
#define NHEAD 4
#define DHEAD 64
#define NC (NNODE * CDIM)

__device__ __forceinline__ v8f wmma16(v16h a, v16h b, v8f c) {
  return __builtin_amdgcn_wmma_f32_16x16x32_f16(false, a, false, b, (short)0, c,
                                                false, false);
}

// Load a 16x32 fragment (A or B role) from an LDS tile.
// lane&15 selects the row (M for A, N for B); K map = (e&7)+16*(e>>3)+8*hi,
// identical for A and B, so the contraction is layout-permutation invariant.
__device__ __forceinline__ v16h frag_ld(const _Float16* base, int row0,
                                        int stride, int col0, int lane) {
  int lm = lane & 15, lhi = lane >> 4;
  const _Float16* p = base + (row0 + lm) * stride + col0 + 8 * lhi;
  union { v16h v; uint4 q[2]; } u;
  u.q[0] = *(const uint4*)(p);
  u.q[1] = *(const uint4*)(p + 16);
  return u.v;
}

// ---------------- generic WMMA GEMM: C = act(A[M,K] @ B[K,N] + bias) --------
// block 256 thr (8 waves), tile 128x64, K step 32. Requires M%128==0, N%64==0,
// K%32==0 (true for all uses here).
__global__ __launch_bounds__(256) void k_gemm(const float* __restrict__ A,
                                              const float* __restrict__ B,
                                              const float* __restrict__ bias,
                                              float* __restrict__ C, int M,
                                              int Nn, int K, int relu) {
  __shared__ alignas(16) _Float16 lA[128 * 40];
  __shared__ alignas(16) _Float16 lB[64 * 40];
  const int tid = threadIdx.x, lane = tid & 31, wid = tid >> 5;
  const int wrow = (wid & 3) * 32, wcol = (wid >> 2) * 32;
  const int mb = blockIdx.y * 128, nb = blockIdx.x * 64;
  v8f a00{}, a01{}, a10{}, a11{};
  for (int k0 = 0; k0 < K; k0 += 32) {
    for (int t = 0; t < 16; ++t) {
      int i = t * 256 + tid;
      int r = i >> 5, c = i & 31;
      lA[r * 40 + c] = (_Float16)A[(size_t)(mb + r) * K + k0 + c];
    }
    for (int t = 0; t < 8; ++t) {
      int i = t * 256 + tid;
      int kk = i >> 6, nn = i & 63;
      lB[nn * 40 + kk] = (_Float16)B[(size_t)(k0 + kk) * Nn + nb + nn];
    }
    __syncthreads();
    v16h af0 = frag_ld(lA, wrow, 40, 0, lane);
    v16h af1 = frag_ld(lA, wrow + 16, 40, 0, lane);
    v16h bf0 = frag_ld(lB, wcol, 40, 0, lane);
    v16h bf1 = frag_ld(lB, wcol + 16, 40, 0, lane);
    a00 = wmma16(af0, bf0, a00);
    a01 = wmma16(af0, bf1, a01);
    a10 = wmma16(af1, bf0, a10);
    a11 = wmma16(af1, bf1, a11);
    __syncthreads();
  }
  const int lm = lane & 15, lhi = lane >> 4;
  v8f accs[2][2] = {{a00, a01}, {a10, a11}};
  for (int fm = 0; fm < 2; ++fm)
    for (int fn = 0; fn < 2; ++fn) {
      int col = nb + wcol + fn * 16 + lm;
      float bv = bias ? bias[col] : 0.f;
      for (int r = 0; r < 8; ++r) {
        int row = mb + wrow + fm * 16 + r + 8 * lhi;
        float v = accs[fm][fn][r] + bv;
        if (relu) v = fmaxf(v, 0.f);
        C[(size_t)row * Nn + col] = v;
      }
    }
}

// ---------------- flash-style MHA: one head per blockIdx.y ------------------
// block = 8 waves, each wave owns 16 queries; stream keys in 32-chunks.
__global__ __launch_bounds__(256) void k_flash(const float* __restrict__ Q,
                                               const float* __restrict__ Km,
                                               const float* __restrict__ Vm,
                                               float* __restrict__ O) {
  __shared__ alignas(16) _Float16 ldsK[32 * 72];    // [key][dim]
  __shared__ alignas(16) _Float16 ldsV[64 * 40];    // [dim][key]
  __shared__ alignas(16) _Float16 ldsP[8][16 * 40]; // per-wave [q][key]
  const int h = blockIdx.y;
  const int tid = threadIdx.x, lane = tid & 31, wid = tid >> 5;
  const int lm = lane & 15, lhi = lane >> 4;
  const int qbase = blockIdx.x * 128 + wid * 16;

  // Q fragments for this wave's 16 queries (dims 0..63 -> two A frags)
  v16h qa[2];
  {
    const float* qrow = Q + (size_t)(qbase + lm) * CDIM + h * DHEAD;
    for (int f = 0; f < 2; ++f) {
      union { v16h v; _Float16 e[16]; } u;
      for (int e2 = 0; e2 < 16; ++e2) {
        int kk = (e2 & 7) + 16 * (e2 >> 3) + 8 * lhi + f * 32;
        u.e[e2] = (_Float16)qrow[kk];
      }
      qa[f] = u.v;
    }
  }

  v8f o0{}, o1{}, o2{}, o3{};
  float mrun[8], lrun[8];
  for (int r = 0; r < 8; ++r) { mrun[r] = -3.0e38f; lrun[r] = 0.f; }

  for (int kc = 0; kc < NNODE; kc += 32) {
    for (int i = tid; i < 32 * 64; i += 256) {
      int key = i >> 6, d = i & 63;
      float kvv = Km[(size_t)(kc + key) * CDIM + h * DHEAD + d];
      float vvv = Vm[(size_t)(kc + key) * CDIM + h * DHEAD + d];
      ldsK[key * 72 + d] = (_Float16)kvv;
      ldsV[d * 40 + key] = (_Float16)vvv;
    }
    __syncthreads();

    v8f s0{}, s1{};
    for (int f = 0; f < 2; ++f) {
      v16h b0 = frag_ld(ldsK, 0, 72, f * 32, lane);
      v16h b1 = frag_ld(ldsK, 16, 72, f * 32, lane);
      s0 = wmma16(qa[f], b0, s0);
      s1 = wmma16(qa[f], b1, s1);
    }
    const float sc = 0.125f; // 1/sqrt(64)
    float mx[8], p0[8], p1[8], scalef[8], ls[8];
    for (int r = 0; r < 8; ++r) {
      float a = s0[r] * sc, b = s1[r] * sc;
      s0[r] = a; s1[r] = b;
      mx[r] = fmaxf(a, b);
    }
    for (int msk = 1; msk < 16; msk <<= 1)
      for (int r = 0; r < 8; ++r) mx[r] = fmaxf(mx[r], __shfl_xor(mx[r], msk, 32));
    for (int r = 0; r < 8; ++r) {
      float mn = fmaxf(mrun[r], mx[r]);
      scalef[r] = __expf(mrun[r] - mn);
      mrun[r] = mn;
      p0[r] = __expf(s0[r] - mn);
      p1[r] = __expf(s1[r] - mn);
      ls[r] = p0[r] + p1[r];
    }
    for (int msk = 1; msk < 16; msk <<= 1)
      for (int r = 0; r < 8; ++r) ls[r] += __shfl_xor(ls[r], msk, 32);
    for (int r = 0; r < 8; ++r) {
      lrun[r] = lrun[r] * scalef[r] + ls[r];
      o0[r] *= scalef[r]; o1[r] *= scalef[r];
      o2[r] *= scalef[r]; o3[r] *= scalef[r];
    }
    _Float16* pw = &ldsP[wid][0];
    for (int r = 0; r < 8; ++r) {
      pw[(r + 8 * lhi) * 40 + lm] = (_Float16)p0[r];
      pw[(r + 8 * lhi) * 40 + 16 + lm] = (_Float16)p1[r];
    }
    __syncthreads();
    v16h pa = frag_ld(pw, 0, 40, 0, lane);
    v16h vb0 = frag_ld(ldsV, 0, 40, 0, lane);
    v16h vb1 = frag_ld(ldsV, 16, 40, 0, lane);
    v16h vb2 = frag_ld(ldsV, 32, 40, 0, lane);
    v16h vb3 = frag_ld(ldsV, 48, 40, 0, lane);
    o0 = wmma16(pa, vb0, o0);
    o1 = wmma16(pa, vb1, o1);
    o2 = wmma16(pa, vb2, o2);
    o3 = wmma16(pa, vb3, o3);
    __syncthreads();
  }
  for (int r = 0; r < 8; ++r) {
    float inv = 1.0f / lrun[r];
    float* dst = O + (size_t)(qbase + r + 8 * lhi) * CDIM + h * DHEAD;
    dst[0 * 16 + lm] = o0[r] * inv;
    dst[1 * 16 + lm] = o1[r] * inv;
    dst[2 * 16 + lm] = o2[r] * inv;
    dst[3 * 16 + lm] = o3[r] * inv;
  }
}

// ---------------- GAT edge kernels ------------------------------------------
__device__ __forceinline__ void atomicMaxF(float* addr, float val) {
  if (val >= 0.f)
    atomicMax((int*)addr, __float_as_int(val));
  else
    atomicMin((unsigned int*)addr, __float_as_uint(val));
}

__device__ __forceinline__ void edge_pair(const int* ei, int Ebase, int eid,
                                          int& src, int& dst) {
  if (eid < Ebase) { src = ei[eid]; dst = ei[Ebase + eid]; }
  else { src = dst = eid - Ebase; } // self loops appended
}

__global__ void k_edge_logits(const float* __restrict__ xs,
                              const float* __restrict__ xd,
                              const int* __restrict__ ei, int Ebase, int Etot,
                              const float* __restrict__ att,
                              float* __restrict__ ebuf,
                              float* __restrict__ mbuf) {
  int t = blockIdx.x * blockDim.x + threadIdx.x;
  if (t >= Etot * NHEAD) return;
  int eid = t >> 2, h = t & 3;
  int src, dst;
  edge_pair(ei, Ebase, eid, src, dst);
  const float* ps = xs + (size_t)src * CDIM + h * DHEAD;
  const float* pd = xd + (size_t)dst * CDIM + h * DHEAD;
  const float* pa = att + h * DHEAD;
  float acc = 0.f;
  for (int d = 0; d < DHEAD; ++d) {
    float v = ps[d] + pd[d];
    v = v > 0.f ? v : 0.2f * v; // leaky_relu(0.2)
    acc += pa[d] * v;
  }
  ebuf[eid * 4 + h] = acc;
  atomicMaxF(mbuf + dst * 4 + h, acc);
}

__global__ void k_edge_exp(const int* __restrict__ ei, int Ebase, int Etot,
                           float* __restrict__ ebuf,
                           const float* __restrict__ mbuf,
                           float* __restrict__ sbuf) {
  int t = blockIdx.x * blockDim.x + threadIdx.x;
  if (t >= Etot * NHEAD) return;
  int eid = t >> 2, h = t & 3;
  int src, dst;
  edge_pair(ei, Ebase, eid, src, dst);
  float p = __expf(ebuf[eid * 4 + h] - mbuf[dst * 4 + h]);
  ebuf[eid * 4 + h] = p;
  atomicAdd(sbuf + dst * 4 + h, p);
}

__global__ void k_edge_aggr(const float* __restrict__ xs,
                            const int* __restrict__ ei, int Ebase,
                            const float* __restrict__ pbuf,
                            const float* __restrict__ sbuf,
                            float* __restrict__ out) {
  size_t t = (size_t)blockIdx.x * blockDim.x + threadIdx.x;
  int eid = (int)(t >> 8);
  int c = (int)(t & 255);
  int h = c >> 6;
  int src, dst;
  edge_pair(ei, Ebase, eid, src, dst);
  float alpha = pbuf[eid * 4 + h] / (sbuf[dst * 4 + h] + 1e-16f);
  atomicAdd(out + (size_t)dst * CDIM + c, alpha * xs[(size_t)src * CDIM + c]);
}

// ---------------- small elementwise kernels ---------------------------------
__global__ void k_fill(float* p, float v, int n) {
  int i = blockIdx.x * blockDim.x + threadIdx.x;
  if (i < n) p[i] = v;
}
__global__ void k_fill_bias(float* p, const float* __restrict__ bias, int n) {
  int i = blockIdx.x * blockDim.x + threadIdx.x;
  if (i < n) p[i] = bias[i & (CDIM - 1)];
}
__global__ void k_relu(float* p, int n) {
  int i = blockIdx.x * blockDim.x + threadIdx.x;
  if (i < n) p[i] = fmaxf(p[i], 0.f);
}

__global__ void k_ln(const float* __restrict__ hb, const float* __restrict__ en,
                     const float* __restrict__ g, const float* __restrict__ b,
                     float* __restrict__ out) {
  __shared__ float red[256];
  int row = blockIdx.x, tid = threadIdx.x;
  float v = hb[(size_t)row * CDIM + tid] + en[(size_t)row * CDIM + tid];
  red[tid] = v;
  __syncthreads();
  for (int s = 128; s > 0; s >>= 1) {
    if (tid < s) red[tid] += red[tid + s];
    __syncthreads();
  }
  float mu = red[0] * (1.0f / CDIM);
  __syncthreads();
  float d = v - mu;
  red[tid] = d * d;
  __syncthreads();
  for (int s = 128; s > 0; s >>= 1) {
    if (tid < s) red[tid] += red[tid + s];
    __syncthreads();
  }
  float var = red[0] * (1.0f / CDIM);
  out[(size_t)row * CDIM + tid] = d * rsqrtf(var + 1e-5f) * g[tid] + b[tid];
}

__global__ void k_cls2(const float* __restrict__ a, const float* __restrict__ w,
                       const float* __restrict__ b2, float* __restrict__ out,
                       int M) {
  int n = blockIdx.x * blockDim.x + threadIdx.x;
  if (n >= M) return;
  const float* ar = a + (size_t)n * 128;
  float acc0 = b2[0], acc1 = b2[1];
  for (int k = 0; k < 128; ++k) {
    float v = ar[k];
    acc0 += v * w[k * 2 + 0];
    acc1 += v * w[k * 2 + 1];
  }
  out[n * 2 + 0] = acc0;
  out[n * 2 + 1] = acc1;
}

// ---------------- host-side orchestration -----------------------------------
extern "C" void kernel_launch(void* const* d_in, const int* in_sizes, int n_in,
                              void* d_out, int out_size, void* d_ws,
                              size_t ws_size, hipStream_t stream) {
  const float* x = (const float*)d_in[0];
  const int* ei = (const int*)d_in[1];
  const float* emb_w = (const float*)d_in[2];
  const float* emb_b = (const float*)d_in[3];
  const float* g1_wl = (const float*)d_in[4];
  const float* g1_wr = (const float*)d_in[5];
  const float* g1_att = (const float*)d_in[6];
  const float* g1_b = (const float*)d_in[7];
  const float* g2_wl = (const float*)d_in[8];
  const float* g2_wr = (const float*)d_in[9];
  const float* g2_att = (const float*)d_in[10];
  const float* g2_b = (const float*)d_in[11];
  const float* wq = (const float*)d_in[12];
  const float* wk = (const float*)d_in[13];
  const float* wv = (const float*)d_in[14];
  const float* bq = (const float*)d_in[15];
  const float* bk = (const float*)d_in[16];
  const float* bv = (const float*)d_in[17];
  const float* wo = (const float*)d_in[18];
  const float* bo = (const float*)d_in[19];
  const float* ln_g = (const float*)d_in[20];
  const float* ln_b = (const float*)d_in[21];
  const float* cw1 = (const float*)d_in[22];
  const float* cb1 = (const float*)d_in[23];
  const float* cw2 = (const float*)d_in[24];
  const float* cb2 = (const float*)d_in[25];

  const int Ebase = in_sizes[1] / 2;
  const int Etot = Ebase + NNODE;

  float* ws = (float*)d_ws;
  float* B_h = ws;                        // [N,C]
  float* B_t1 = B_h + NC;                 // [N,C]
  float* B_t2 = B_t1 + NC;                // [N,C]
  float* B_t3 = B_t2 + NC;                // [N,C]
  float* B_t4 = B_t3 + NC;                // [N,C]
  float* B_e = B_t4 + NC;                 // [Etot,4]
  float* B_m = B_e + (size_t)Etot * 4;    // [N,4]
  float* B_s = B_m + NNODE * 4;           // [N,4]
  float* B_c1 = B_s + NNODE * 4;          // [N,128]

  const dim3 gemmB(256);
  const dim3 gC(CDIM / 64, NNODE / 128);  // N=256 GEMMs
  const dim3 gC1(128 / 64, NNODE / 128);  // N=128 GEMM
  const int eT = (Etot * 4 + 255) / 256;

  // 1) embedding
  k_gemm<<<gC, gemmB, 0, stream>>>(x, emb_w, emb_b, B_h, NNODE, CDIM, 768, 0);

  // 2) GAT layer 1: B_h -> B_t3
  k_gemm<<<gC, gemmB, 0, stream>>>(B_h, g1_wl, nullptr, B_t1, NNODE, CDIM, CDIM, 0);
  k_gemm<<<gC, gemmB, 0, stream>>>(B_h, g1_wr, nullptr, B_t2, NNODE, CDIM, CDIM, 0);
  k_fill<<<(NNODE * 4 + 255) / 256, 256, 0, stream>>>(B_m, -3.0e38f, NNODE * 4);
  k_fill<<<(NNODE * 4 + 255) / 256, 256, 0, stream>>>(B_s, 0.f, NNODE * 4);
  k_fill_bias<<<(NC + 255) / 256, 256, 0, stream>>>(B_t3, g1_b, NC);
  k_edge_logits<<<eT, 256, 0, stream>>>(B_t1, B_t2, ei, Ebase, Etot, g1_att, B_e, B_m);
  k_edge_exp<<<eT, 256, 0, stream>>>(ei, Ebase, Etot, B_e, B_m, B_s);
  k_edge_aggr<<<Etot, 256, 0, stream>>>(B_t1, ei, Ebase, B_e, B_s, B_t3);
  k_relu<<<(NC + 255) / 256, 256, 0, stream>>>(B_t3, NC);

  // 3) GAT layer 2: B_t3 -> B_h
  k_gemm<<<gC, gemmB, 0, stream>>>(B_t3, g2_wl, nullptr, B_t1, NNODE, CDIM, CDIM, 0);
  k_gemm<<<gC, gemmB, 0, stream>>>(B_t3, g2_wr, nullptr, B_t2, NNODE, CDIM, CDIM, 0);
  k_fill<<<(NNODE * 4 + 255) / 256, 256, 0, stream>>>(B_m, -3.0e38f, NNODE * 4);
  k_fill<<<(NNODE * 4 + 255) / 256, 256, 0, stream>>>(B_s, 0.f, NNODE * 4);
  k_fill_bias<<<(NC + 255) / 256, 256, 0, stream>>>(B_h, g2_b, NC);
  k_edge_logits<<<eT, 256, 0, stream>>>(B_t1, B_t2, ei, Ebase, Etot, g2_att, B_e, B_m);
  k_edge_exp<<<eT, 256, 0, stream>>>(ei, Ebase, Etot, B_e, B_m, B_s);
  k_edge_aggr<<<Etot, 256, 0, stream>>>(B_t1, ei, Ebase, B_e, B_s, B_h);
  k_relu<<<(NC + 255) / 256, 256, 0, stream>>>(B_h, NC);

  // 4) MHA: q/k/v, flash attention, output projection
  k_gemm<<<gC, gemmB, 0, stream>>>(B_h, wq, bq, B_t1, NNODE, CDIM, CDIM, 0);
  k_gemm<<<gC, gemmB, 0, stream>>>(B_h, wk, bk, B_t2, NNODE, CDIM, CDIM, 0);
  k_gemm<<<gC, gemmB, 0, stream>>>(B_h, wv, bv, B_t3, NNODE, CDIM, CDIM, 0);
  k_flash<<<dim3(NNODE / 128, NHEAD), 256, 0, stream>>>(B_t1, B_t2, B_t3, B_t4);
  k_gemm<<<gC, gemmB, 0, stream>>>(B_t4, wo, bo, B_t1, NNODE, CDIM, CDIM, 0);

  // 5) residual + LayerNorm -> B_t2
  k_ln<<<NNODE, 256, 0, stream>>>(B_h, B_t1, ln_g, ln_b, B_t2);

  // 6) classifier
  k_gemm<<<gC1, gemmB, 0, stream>>>(B_t2, cw1, cb1, B_c1, NNODE, 128, CDIM, 1);
  k_cls2<<<(NNODE + 255) / 256, 256, 0, stream>>>(B_c1, cw2, cb2, (float*)d_out,
                                                  NNODE);
}
